// GCN_with_dense_21131239097136
// MI455X (gfx1250) — compile-verified, matrix operand order checked by reading, not run on hardware
//
#include <hip/hip_runtime.h>
#include <hip/hip_bf16.h>

typedef __attribute__((ext_vector_type(16))) _Float16 v16h;
typedef __attribute__((ext_vector_type(8)))  float    v8f;

#define N_NODES 100000
#define N_EDGES 1600000
#define N_GRAPH 1024
#define N_CLASS 104

// ------------------------------------------------------------------
// zero fill (float4 granularity; all our buffers are multiples of 4)
// ------------------------------------------------------------------
__global__ void zero_kernel(float4* __restrict__ p, int n4) {
  int i = blockIdx.x * blockDim.x + threadIdx.x;
  if (i < n4) p[i] = make_float4(0.f, 0.f, 0.f, 0.f);
}

// ------------------------------------------------------------------
// WMMA GEMM: out[M,Ncols] = op(A[M,K]) @ W[K,Ncols] (+bias)
//   - KT compile-time -> k-loop fully unrolled, back-to-back v_wmma
//   - W staged f32->f16 into LDS once per block (cols zero-padded to Npad)
//   - each wave owns one 16-wide column strip (tileN); its KT/32 B
//     fragments are pulled from LDS into registers ONCE, then the wave
//     grid-strides over M tiles doing only A loads + WMMA (no per-tile DS)
//   - RELU_A fused into A-fragment load; bias fused into store
// Fragment layouts per CDNA5 ISA §7.12.2 (wave32):
//   A 16x32 f16: lane m=L&15, hs=L>>4; elem e -> K = e + (e<8?0:8) + 8*hs
//   B 32x16 f16: lane n=L&15, same per-element K mapping
//   C/D 16x16 f32: lane col n=L&15, VGPR r -> row = 8*hs + r
// ------------------------------------------------------------------
template<int KT, bool RELU_A, bool ADD_BIAS>
__global__ void __launch_bounds__(128)
gemm_wmma(const float* __restrict__ A, const float* __restrict__ W,
          const float* __restrict__ bias, float* __restrict__ out,
          int M, int Ncols, int Npad)
{
  __shared__ _Float16 sW[7168];   // max K*Npad = 64*112 halves = 14 KB
  const int total = KT * Npad;
  for (int idx = threadIdx.x; idx < total; idx += blockDim.x) {
    int k = idx / Npad, n = idx - k * Npad;
    float v = (n < Ncols) ? W[(size_t)k * Ncols + n] : 0.f;
    sW[idx] = (_Float16)v;
  }
  __syncthreads();

  const int wavesTotal = gridDim.x * (blockDim.x >> 5);
  const int wid = blockIdx.x * (blockDim.x >> 5) + (threadIdx.x >> 5);
  const int ntiles = Npad >> 4;
  const int wavesPerN = wavesTotal / ntiles;   // launch guarantees divisibility
  const int tileN = wid % ntiles;
  const int mstart = wid / ntiles;
  const int mTiles = M >> 4;                   // M is a multiple of 16 here

  const int lane = threadIdx.x & 31;
  const int mr = lane & 15;
  const int hs = lane >> 4;
  const int ncol = tileN * 16 + mr;

  // ---- load this wave's B fragments from LDS into registers (once) ----
  constexpr int KSTEPS = KT / 32;
  v16h bfrag[KSTEPS];
#pragma unroll
  for (int ks = 0; ks < KSTEPS; ++ks) {
    const int kb = ks * 32 + 8 * hs;
#pragma unroll
    for (int i = 0; i < 8; ++i) {
      bfrag[ks][i]     = sW[(kb + i) * Npad + ncol];
      bfrag[ks][i + 8] = sW[(kb + 16 + i) * Npad + ncol];
    }
  }

  const float bv = ADD_BIAS ? ((ncol < Ncols) ? bias[ncol] : 0.f) : 0.f;

  // ---- persistent loop over M tiles: A loads + unrolled WMMA chain ----
  for (int tm = mstart; tm < mTiles; tm += wavesPerN) {
    const float* __restrict__ arow = A + (size_t)(tm * 16 + mr) * KT;
    v8f acc = {};
#pragma unroll
    for (int ks = 0; ks < KSTEPS; ++ks) {
      const int kb = ks * 32 + 8 * hs;
      v16h a;
#pragma unroll
      for (int i = 0; i < 8; ++i) {
        float v0 = arow[kb + i];
        float v1 = arow[kb + 16 + i];
        if (RELU_A) { v0 = fmaxf(v0, 0.f); v1 = fmaxf(v1, 0.f); }
        a[i]     = (_Float16)v0;
        a[i + 8] = (_Float16)v1;
      }
      acc = __builtin_amdgcn_wmma_f32_16x16x32_f16(
          /*neg_a=*/false, a, /*neg_b=*/false, bfrag[ks],
          /*c_mod=*/(short)0, acc, /*reuse_a=*/false, /*reuse_b=*/false);
    }
    if (ncol < Ncols) {                        // per-lane only AFTER the WMMAs
#pragma unroll
      for (int r = 0; r < 8; ++r) {
        int row = tm * 16 + 8 * hs + r;
        out[(size_t)row * Ncols + ncol] = acc[r] + bv;
      }
    }
  }
}

// ------------------------------------------------------------------
// SpMM scatter: out[dst[e],:] += w[e] * y[src[e],:]
// one thread per (edge, 4-feature chunk); fshift = log2(Fc/4)
// Feature matrices fit in L2 (<=25.6MB of 192MB) so the random
// gathers + float atomics stay on-chip.
// ------------------------------------------------------------------
__global__ void spmm_kernel(const float* __restrict__ y, const int* __restrict__ src,
                            const int* __restrict__ dst, const float* __restrict__ w,
                            float* __restrict__ out, int E, int fshift)
{
  int gid = blockIdx.x * blockDim.x + threadIdx.x;
  int e = gid >> fshift;
  if (e >= E) return;
  int f4 = gid & ((1 << fshift) - 1);
  if (f4 == 0 && e + 4096 < E) {              // warm the edge-index stream
    __builtin_prefetch(&src[e + 4096], 0, 0);
    __builtin_prefetch(&dst[e + 4096], 0, 0);
  }
  int s = src[e], d = dst[e];
  float ww = w[e];
  float4 v = ((const float4*)y)[((size_t)s << fshift) + f4];
  float* o = out + ((((size_t)d << fshift) + f4) << 2);
  atomicAdd(o + 0, v.x * ww);
  atomicAdd(o + 1, v.y * ww);
  atomicAdd(o + 2, v.z * ww);
  atomicAdd(o + 3, v.w * ww);
}

// ------------------------------------------------------------------
// pooled[seg[n],:] += pw[n] * relu(h[n,:]);  Fc = 64
// ------------------------------------------------------------------
__global__ void pool_kernel(const float* __restrict__ h, const int* __restrict__ seg,
                            const float* __restrict__ pw, float* __restrict__ pooled, int n)
{
  int gid = blockIdx.x * blockDim.x + threadIdx.x;
  int node = gid >> 4;
  if (node >= n) return;
  int f4 = gid & 15;
  int g = seg[node];
  float ww = pw[node];
  float4 v = ((const float4*)h)[(size_t)node * 16 + f4];
  float* o = pooled + (size_t)g * 64 + f4 * 4;
  atomicAdd(o + 0, fmaxf(v.x, 0.f) * ww);
  atomicAdd(o + 1, fmaxf(v.y, 0.f) * ww);
  atomicAdd(o + 2, fmaxf(v.z, 0.f) * ww);
  atomicAdd(o + 3, fmaxf(v.w, 0.f) * ww);
}

// ------------------------------------------------------------------
// loss = 0.00025 * sum(W1^2) + mean_g( logsumexp(logits_g) - logits_g[label_g] )
// ------------------------------------------------------------------
__global__ void __launch_bounds__(1024)
loss_kernel(const float* __restrict__ logits, const int* __restrict__ label,
            const float* __restrict__ W1, float* __restrict__ loss_out)
{
  __shared__ float red[1024];
  int t = threadIdx.x;
  const float* row = logits + (size_t)t * N_CLASS;
  float mx = -3.402823466e38f;
  for (int j = 0; j < N_CLASS; ++j) mx = fmaxf(mx, row[j]);
  float s = 0.f;
  for (int j = 0; j < N_CLASS; ++j) s += __expf(row[j] - mx);
  float ce = mx + __logf(s) - row[label[t]];
  red[t] = ce;
  __syncthreads();
  for (int off = 512; off > 0; off >>= 1) {
    if (t < off) red[t] += red[t + off];
    __syncthreads();
  }
  float ce_sum = red[0];
  __syncthreads();
  float wsum = 0.f;
  for (int i = t; i < 128 * 32; i += 1024) { float v = W1[i]; wsum += v * v; }
  red[t] = wsum;
  __syncthreads();
  for (int off = 512; off > 0; off >>= 1) {
    if (t < off) red[t] += red[t + off];
    __syncthreads();
  }
  if (t == 0) loss_out[0] = 0.00025f * red[0] + ce_sum * (1.f / N_GRAPH);
}

// ------------------------------------------------------------------
extern "C" void kernel_launch(void* const* d_in, const int* in_sizes, int n_in,
                              void* d_out, int out_size, void* d_ws, size_t ws_size,
                              hipStream_t stream) {
  (void)in_sizes; (void)n_in; (void)out_size; (void)ws_size;
  const float* x    = (const float*)d_in[0];
  const int*   esrc = (const int*)  d_in[1];
  const int*   edst = (const int*)  d_in[2];
  const float* ew   = (const float*)d_in[3];
  const int*   pseg = (const int*)  d_in[4];
  const float* pw   = (const float*)d_in[5];
  const int*   lbl  = (const int*)  d_in[6];
  const float* W1   = (const float*)d_in[7];
  const float* W2   = (const float*)d_in[8];
  const float* W3   = (const float*)d_in[9];
  const float* Wd   = (const float*)d_in[10];
  const float* bd   = (const float*)d_in[11];
  float* logits = (float*)d_out;

  // workspace: bufP (dense product) | bufH (spmm accum) | bufPool
  float* bufP    = (float*)d_ws;                       // N*64 f32
  float* bufH    = bufP + (size_t)N_NODES * 64;        // N*64 f32
  float* bufPool = bufH + (size_t)N_NODES * 64;        // G*64 f32

  // blocks chosen so total waves = ntiles * wavesPerN (exact divisibility)
  // GEMM1: ntiles=2, WPN=1024 -> 2048 waves -> 512 blocks
  // GEMM2/3: ntiles=4, WPN=512 -> 2048 waves -> 512 blocks
  // head: ntiles=7, WPN=64 -> 448 waves -> 112 blocks

  // ---- layer 1: P = X @ W1   [100000,128]x[128,32]
  gemm_wmma<128, false, false><<<512, 128, 0, stream>>>(x, W1, nullptr, bufP,
                                                        N_NODES, 32, 32);
  { int n4 = N_NODES * 32 / 4;
    zero_kernel<<<(n4 + 255) / 256, 256, 0, stream>>>((float4*)bufH, n4); }
  { int thr = N_EDGES * 8;  // Fc=32 -> 8 float4 chunks/edge
    spmm_kernel<<<(thr + 255) / 256, 256, 0, stream>>>(bufP, esrc, edst, ew, bufH,
                                                       N_EDGES, 3); }

  // ---- layer 2: P = relu(H) @ W2   [100000,32]x[32,64]
  gemm_wmma<32, true, false><<<512, 128, 0, stream>>>(bufH, W2, nullptr, bufP,
                                                      N_NODES, 64, 64);
  { int n4 = N_NODES * 64 / 4;
    zero_kernel<<<(n4 + 255) / 256, 256, 0, stream>>>((float4*)bufH, n4); }
  { int thr = N_EDGES * 16; // Fc=64
    spmm_kernel<<<(thr + 255) / 256, 256, 0, stream>>>(bufP, esrc, edst, ew, bufH,
                                                       N_EDGES, 4); }

  // ---- layer 3: P = relu(H) @ W3   [100000,64]x[64,64]
  gemm_wmma<64, true, false><<<512, 128, 0, stream>>>(bufH, W3, nullptr, bufP,
                                                      N_NODES, 64, 64);
  { int n4 = N_NODES * 64 / 4;
    zero_kernel<<<(n4 + 255) / 256, 256, 0, stream>>>((float4*)bufH, n4); }
  { int thr = N_EDGES * 16;
    spmm_kernel<<<(thr + 255) / 256, 256, 0, stream>>>(bufP, esrc, edst, ew, bufH,
                                                       N_EDGES, 4); }

  // ---- pooling: pooled = segsum(relu(H) * pw)   [1024,64]
  { int n4 = N_GRAPH * 64 / 4;
    zero_kernel<<<(n4 + 255) / 256, 256, 0, stream>>>((float4*)bufPool, n4); }
  { int thr = N_NODES * 16;
    pool_kernel<<<(thr + 255) / 256, 256, 0, stream>>>(bufH, pseg, pw, bufPool, N_NODES); }

  // ---- head: logits = pooled @ Wd + bd   [1024,64]x[64,104] (pad N to 112)
  gemm_wmma<64, false, true><<<112, 128, 0, stream>>>(bufPool, Wd, bd, logits,
                                                      N_GRAPH, N_CLASS, 112);

  // ---- loss -> d_out[G*C]
  loss_kernel<<<1, 1024, 0, stream>>>(logits, lbl, W1, logits + (size_t)N_GRAPH * N_CLASS);
}
